// AttentionLayer_42502996361620
// MI455X (gfx1250) — matrix operand order repaired and checked by reading, not compiled, not run on hardware
//
#include <hip/hip_runtime.h>
#include <hip/hip_bf16.h>
#include <math.h>

// ---------------------------------------------------------------------------
// Transformer block (B=2,S=2048,D=1024,H=8,DQK=128,DFF=4096) on gfx1250.
// All GEMMs via v_wmma_f32_16x16x32_bf16, double-buffered LDS staging,
// software-pipelined global->LDS copies, weights pre-transposed to [K,N].
// Pipeline:
//   1) x -> bf16; all weights -> bf16 transposed ([K,N])
//   2) Q proj; K proj stored transposed (kt[b,h,e,s]); V proj
//   3) scores = Q kt * 1/sqrt(128) (causal tiles skipped), bf16
//   4) causal softmax in-place (zeros above diagonal)
//   5) res1 = x + sum_h P V       (f32 + bf16 copies)
//   6) h    = relu(res1 W1t + b1) (bf16)
//   7) out  = res1 + h W2t + b2   (f32 -> d_out)
// ---------------------------------------------------------------------------

#define TB 2
#define TS 2048
#define TD 1024
#define TH 8
#define TDQK 128
#define TDFF 4096

typedef __attribute__((ext_vector_type(16))) __bf16 v16bf;
typedef __attribute__((ext_vector_type(8)))  float  v8f;

__device__ __forceinline__ unsigned short f32_to_bf16(float f) {
  unsigned u = __builtin_bit_cast(unsigned, f);
  u += 0x7FFFu + ((u >> 16) & 1u);          // round-to-nearest-even
  return (unsigned short)(u >> 16);
}
__device__ __forceinline__ float bf16_to_f32(unsigned short h) {
  unsigned u = ((unsigned)h) << 16;
  return __builtin_bit_cast(float, u);
}

// ------------------------------- f32 -> bf16 -------------------------------
__global__ __launch_bounds__(256) void cvt_f32_bf16(const float* __restrict__ in,
                                                    unsigned short* __restrict__ out,
                                                    long long n) {
  long long i = (long long)blockIdx.x * blockDim.x + threadIdx.x;
  long long stride = (long long)gridDim.x * blockDim.x;
  for (; i < n; i += stride) out[i] = f32_to_bf16(in[i]);
}

// --------------------- f32 [Z,R,C] -> bf16 [Z,C,R] transpose ---------------
__global__ __launch_bounds__(256) void transpose_cvt(const float* __restrict__ in,
                                                     unsigned short* __restrict__ out,
                                                     int R, int C) {
  __shared__ float t[32][33];
  const float* src = in + (size_t)blockIdx.z * R * C;
  unsigned short* dst = out + (size_t)blockIdx.z * R * C;
  const int c0 = blockIdx.x * 32, r0 = blockIdx.y * 32;
  const int tx = threadIdx.x, ty = threadIdx.y;  // 32 x 8
#pragma unroll
  for (int j = 0; j < 32; j += 8)
    t[ty + j][tx] = src[(size_t)(r0 + ty + j) * C + c0 + tx];
  __syncthreads();
#pragma unroll
  for (int j = 0; j < 32; j += 8)
    dst[(size_t)(c0 + ty + j) * R + r0 + tx] = f32_to_bf16(t[tx][ty + j]);
}

// ------------------------------ generic GEMM -------------------------------
// C[M,N] = sum_sb A[M,K] * B[K,N]; A,B bf16; f32 accumulate via WMMA.
struct GemmParams {
  const unsigned short* A;
  const unsigned short* Bm;
  const float* bias;       // [N] or null
  const float* aux;        // f32 tensor indexed like C, or null
  float* outF;             // f32 output or null
  unsigned short* outB;    // bf16 output or null
  long long aOuter, aInner, aSum;
  long long bOuter, bInner, bSum;
  long long cOuter, cInner;
  int M, N, K, lda, ldb;
  int zDiv;                // blockIdx.z -> outer = z/zDiv, inner = z%zDiv
  int sumCount;            // reduction over batches (head sum), >=1
  int causalSkip;          // skip tiles fully above diagonal (scores)
  int kCausal;             // limit K loop to m0+128 (P*V)
  int epi;                 // 0 bf16 | 1 scale->bf16 | 2 +aux->f32&bf16
                           // 3 bias+relu->bf16 | 4 bias+aux->f32
                           // 5 bf16 transposed store (C^T, ld=M)
  float scale;
};

__global__ __launch_bounds__(256, 1) void gemm_wmma_bf16(GemmParams p) {
  const int m0 = blockIdx.y * 128;
  const int n0 = blockIdx.x * 128;
  if (p.causalSkip && (int)blockIdx.x > (int)blockIdx.y) return;  // fully masked

  const int outer = blockIdx.z / p.zDiv;
  const int inner = blockIdx.z % p.zDiv;
  const long long aBase = (long long)outer * p.aOuter + (long long)inner * p.aInner;
  const long long bBase = (long long)outer * p.bOuter + (long long)inner * p.bInner;
  const long long cBase = (long long)outer * p.cOuter + (long long)inner * p.cInner;

  __shared__ unsigned short As[2][128 * 32];   // [m][k], k contiguous
  __shared__ unsigned short Bs[2][32 * 128];   // [k][n], n contiguous

  const int tid  = threadIdx.x;
  const int lane = tid & 31;
  const int wv   = tid >> 5;                   // 8 waves: 4 along M x 2 along N
  const int wm   = (wv & 3) * 32;
  const int wn   = (wv >> 2) * 64;
  const int lm   = lane & 15;
  const int lh   = lane >> 4;

  // per-thread staging slots (2 A chunks + 2 B chunks of 8 bf16 each)
  const int rowA0 = tid >> 2,        kcA0 = (tid & 3) << 3;
  const int rowA1 = (tid + 256) >> 2, kcA1 = (tid & 3) << 3;
  const int ldsA0 = rowA0 * 32 + kcA0, ldsA1 = rowA1 * 32 + kcA1;
  const int krB0 = tid >> 4,         ncB0 = (tid & 15) << 3;
  const int ldsB0 = krB0 * 128 + ncB0, ldsB1 = ldsB0 + 16 * 128;

  v8f acc[2][4];
#pragma unroll
  for (int mi = 0; mi < 2; ++mi)
#pragma unroll
    for (int ni = 0; ni < 4; ++ni)
#pragma unroll
      for (int r = 0; r < 8; ++r) acc[mi][ni][r] = 0.0f;

  const int kLimit = p.kCausal ? min(p.K, m0 + 128) : p.K;
  const int nSteps = kLimit >> 5;
  const size_t bStep = (size_t)32 * p.ldb;

  for (int sb = 0; sb < p.sumCount; ++sb) {
    const unsigned short* Ag = p.A  + aBase + (long long)sb * p.aSum;
    const unsigned short* Bg = p.Bm + bBase + (long long)sb * p.bSum;
    const unsigned short* aP0 = Ag + (size_t)(m0 + rowA0) * p.lda + kcA0;
    const unsigned short* aP1 = Ag + (size_t)(m0 + rowA1) * p.lda + kcA1;
    const unsigned short* bP0 = Bg + (size_t)krB0 * p.ldb + n0 + ncB0;
    const unsigned short* bP1 = bP0 + (size_t)16 * p.ldb;

    if (sb) __syncthreads();  // protect LDS reuse across sum batches

    // prologue: fill buffer 0
    uint4 ra0 = *(const uint4*)aP0, ra1 = *(const uint4*)aP1;
    uint4 rb0 = *(const uint4*)bP0, rb1 = *(const uint4*)bP1;
    aP0 += 32; aP1 += 32; bP0 += bStep; bP1 += bStep;
    *(uint4*)&As[0][ldsA0] = ra0; *(uint4*)&As[0][ldsA1] = ra1;
    *(uint4*)&Bs[0][ldsB0] = rb0; *(uint4*)&Bs[0][ldsB1] = rb1;
    __syncthreads();

    int buf = 0;
    for (int st = 0; st < nSteps; ++st) {
      const bool hasNext = (st + 1 < nSteps);
      if (hasNext) {  // issue next tile's global loads before computing
        ra0 = *(const uint4*)aP0; ra1 = *(const uint4*)aP1;
        rb0 = *(const uint4*)bP0; rb1 = *(const uint4*)bP1;
        aP0 += 32; aP1 += 32; bP0 += bStep; bP1 += bStep;
      }

      // ---- fragments per ISA layouts + 8 WMMAs ----
      union Frag { unsigned u[8]; v16bf v; } a[2], b[4];
#pragma unroll
      for (int mi = 0; mi < 2; ++mi) {
        const int row = wm + mi * 16 + lm;
#pragma unroll
        for (int v = 0; v < 8; ++v) {
          const int kk = 2 * v + ((v >= 4) ? 8 : 0) + 8 * lh;  // 16-bit A 16x32
          a[mi].u[v] = *(const unsigned*)&As[buf][row * 32 + kk];
        }
      }
#pragma unroll
      for (int ni = 0; ni < 4; ++ni) {
        const int nb = wn + ni * 16;
#pragma unroll
        for (int v = 0; v < 8; ++v) {          // lanes -> N, VGPR v = K pair (same k order as A)
          const int kb = 2 * v + ((v >= 4) ? 8 : 0) + 8 * lh;
          b[ni].u[v] = (unsigned)Bs[buf][kb * 128 + nb + lm] |
                       ((unsigned)Bs[buf][(kb + 1) * 128 + nb + lm] << 16);
        }
      }
#pragma unroll
      for (int mi = 0; mi < 2; ++mi)
#pragma unroll
        for (int ni = 0; ni < 4; ++ni)
          acc[mi][ni] = __builtin_amdgcn_wmma_f32_16x16x32_bf16(
              false, a[mi].v, false, b[ni].v, (short)0, acc[mi][ni], false, false);

      if (hasNext) {  // commit next tile to the other buffer
        const int nb2 = buf ^ 1;
        *(uint4*)&As[nb2][ldsA0] = ra0; *(uint4*)&As[nb2][ldsA1] = ra1;
        *(uint4*)&Bs[nb2][ldsB0] = rb0; *(uint4*)&Bs[nb2][ldsB1] = rb1;
        __syncthreads();
        buf = nb2;
      }
    }
  }

  // ---- epilogue ----
#pragma unroll
  for (int mi = 0; mi < 2; ++mi)
#pragma unroll
    for (int ni = 0; ni < 4; ++ni)
#pragma unroll
      for (int r = 0; r < 8; ++r) {
        const int mg = m0 + wm + mi * 16 + r + 8 * lh;  // C: VGPR r -> M=r / r+8
        const int ng = n0 + wn + ni * 16 + lm;          // lanes -> N
        if (mg >= p.M || ng >= p.N) continue;
        const float v = acc[mi][ni][r];
        const size_t c = (size_t)cBase + (size_t)mg * p.N + ng;
        switch (p.epi) {
          case 0: p.outB[c] = f32_to_bf16(v); break;
          case 1: p.outB[c] = f32_to_bf16(v * p.scale); break;
          case 2: {
            const float r2 = v + p.aux[c];
            p.outF[c] = r2;
            p.outB[c] = f32_to_bf16(r2);
          } break;
          case 3: p.outB[c] = f32_to_bf16(fmaxf(v + p.bias[ng], 0.0f)); break;
          case 4: p.outF[c] = v + p.bias[ng] + p.aux[c]; break;
          case 5:  // transposed bf16 store: C^T with leading dim M
            p.outB[(size_t)cBase + (size_t)ng * p.M + mg] = f32_to_bf16(v);
            break;
        }
      }
}

// ---------------------- causal softmax, in-place on bf16 -------------------
// One block per row of P[B*H*S, S]. Masked entries (t>s) never read; written 0.
__global__ __launch_bounds__(256) void softmax_causal_inplace(unsigned short* P) {
  const int row = blockIdx.x;
  const int s   = row % TS;
  unsigned short* Pr = P + (size_t)row * TS;
  const int tid = threadIdx.x;
  __shared__ float sm[256];

  float loc[TS / 256];
  float mx = -3.0e38f;
#pragma unroll
  for (int i = 0; i < TS / 256; ++i) {
    int t = tid + i * 256;
    float v = (t <= s) ? bf16_to_f32(Pr[t]) : -3.0e38f;
    loc[i] = v;
    mx = fmaxf(mx, v);
  }
  sm[tid] = mx; __syncthreads();
  for (int o = 128; o > 0; o >>= 1) {
    if (tid < o) sm[tid] = fmaxf(sm[tid], sm[tid + o]);
    __syncthreads();
  }
  mx = sm[0]; __syncthreads();

  float sum = 0.0f;
#pragma unroll
  for (int i = 0; i < TS / 256; ++i) {
    int t = tid + i * 256;
    float e = (t <= s) ? __expf(loc[i] - mx) : 0.0f;
    loc[i] = e;
    sum += e;
  }
  sm[tid] = sum; __syncthreads();
  for (int o = 128; o > 0; o >>= 1) {
    if (tid < o) sm[tid] += sm[tid + o];
    __syncthreads();
  }
  float inv = 1.0f / sm[0];
#pragma unroll
  for (int i = 0; i < TS / 256; ++i)
    Pr[tid + i * 256] = f32_to_bf16(loc[i] * inv);
}

// ---------------------------------------------------------------------------
extern "C" void kernel_launch(void* const* d_in, const int* in_sizes, int n_in,
                              void* d_out, int out_size, void* d_ws, size_t ws_size,
                              hipStream_t stream) {
  const float* x  = (const float*)d_in[0];
  const float* Wq = (const float*)d_in[1];
  const float* Wk = (const float*)d_in[2];
  const float* Wv = (const float*)d_in[3];
  const float* W1 = (const float*)d_in[4];
  const float* b1 = (const float*)d_in[5];
  const float* W2 = (const float*)d_in[6];
  const float* b2 = (const float*)d_in[7];
  // d_in[8] = mask: causality applied analytically, mask unused.
  float* out = (float*)d_out;

  char* ws = (char*)d_ws;
  size_t off = 0;
  auto alloc = [&](size_t bytes) { size_t o = off; off += (bytes + 255) & ~(size_t)255; return o; };

  unsigned short* xb   = (unsigned short*)(ws + alloc((size_t)TB * TS * TD * 2));
  unsigned short* wqt  = (unsigned short*)(ws + alloc((size_t)TH * TD * TDQK * 2));  // [h,D,DQK]
  unsigned short* wkt  = (unsigned short*)(ws + alloc((size_t)TH * TD * TDQK * 2));  // [h,D,DQK]
  unsigned short* wvt  = (unsigned short*)(ws + alloc((size_t)TH * TD * TD * 2));    // [h,D,D]
  unsigned short* w1t  = (unsigned short*)(ws + alloc((size_t)TD * TDFF * 2));       // [D,DFF]
  unsigned short* w2t  = (unsigned short*)(ws + alloc((size_t)TDFF * TD * 2));       // [DFF,D]
  unsigned short* qb   = (unsigned short*)(ws + alloc((size_t)TB * TH * TS * TDQK * 2));
  unsigned short* kt   = (unsigned short*)(ws + alloc((size_t)TB * TH * TDQK * TS * 2));
  unsigned short* vb   = (unsigned short*)(ws + alloc((size_t)TB * TH * TS * TD * 2));
  unsigned short* pb   = (unsigned short*)(ws + alloc((size_t)TB * TH * TS * TS * 2));
  float*          res1 = (float*)         (ws + alloc((size_t)TB * TS * TD * 4));
  unsigned short* res1b= (unsigned short*)(ws + alloc((size_t)TB * TS * TD * 2));
  unsigned short* hb   = (unsigned short*)(ws + alloc((size_t)TB * TS * TDFF * 2));
  (void)ws_size; (void)in_sizes; (void)n_in; (void)out_size;

  // 1) conversions / weight transposes
  cvt_f32_bf16<<<1024, 256, 0, stream>>>(x, xb, (long long)TB * TS * TD);
  dim3 tb(32, 8);
  transpose_cvt<<<dim3(TD / 32, TDQK / 32, TH), tb, 0, stream>>>(Wq, wqt, TDQK, TD);
  transpose_cvt<<<dim3(TD / 32, TDQK / 32, TH), tb, 0, stream>>>(Wk, wkt, TDQK, TD);
  transpose_cvt<<<dim3(TD / 32, TD / 32, TH),   tb, 0, stream>>>(Wv, wvt, TD, TD);
  transpose_cvt<<<dim3(TD / 32, TDFF / 32, 1),  tb, 0, stream>>>(W1, w1t, TDFF, TD);
  transpose_cvt<<<dim3(TDFF / 32, TD / 32, 1),  tb, 0, stream>>>(W2, w2t, TD, TDFF);

  auto gemm = [&](GemmParams p, int Z) {
    dim3 grid((p.N + 127) / 128, (p.M + 127) / 128, Z);
    gemm_wmma_bf16<<<grid, 256, 0, stream>>>(p);
  };

  GemmParams p{};
  // 2) Q projection: q[b,h] = x[b] (SxD) * wqt[h] (DxDQK)
  p = GemmParams{};
  p.A = xb; p.Bm = wqt; p.outB = qb;
  p.M = TS; p.N = TDQK; p.K = TD; p.lda = TD; p.ldb = TDQK;
  p.zDiv = TH;
  p.aOuter = (long long)TS * TD;
  p.bOuter = 0; p.bInner = (long long)TD * TDQK;
  p.cOuter = (long long)TH * TS * TDQK; p.cInner = (long long)TS * TDQK;
  p.sumCount = 1; p.epi = 0;
  gemm(p, TB * TH);

  //    K projection, stored transposed: kt[b,h,e,s]
  p.Bm = wkt; p.outB = kt;
  p.cOuter = (long long)TH * TDQK * TS; p.cInner = (long long)TDQK * TS;
  p.epi = 5;
  gemm(p, TB * TH);

  //    V projection: v[b,h] = x[b] * wvt[h] (DxD)
  p.Bm = wvt; p.outB = vb;
  p.N = TD; p.ldb = TD; p.bInner = (long long)TD * TD;
  p.cOuter = (long long)TH * TS * TD; p.cInner = (long long)TS * TD;
  p.epi = 0;
  gemm(p, TB * TH);

  // 3) scores: P = q kt / sqrt(DQK), causal tiles skipped, bf16
  p = GemmParams{};
  p.A = qb; p.Bm = kt; p.outB = pb;
  p.M = TS; p.N = TS; p.K = TDQK; p.lda = TDQK; p.ldb = TS;
  p.zDiv = 1;
  p.aOuter = (long long)TS * TDQK; p.bOuter = (long long)TDQK * TS;
  p.cOuter = (long long)TS * TS;
  p.sumCount = 1; p.causalSkip = 1; p.epi = 1;
  p.scale = 0.08838834764831845f;   // 1/sqrt(128)
  gemm(p, TB * TH);

  // 4) causal softmax in place
  softmax_causal_inplace<<<TB * TH * TS, 256, 0, stream>>>(pb);

  // 5) res1 = x + sum_h P[b,h] V[b,h]
  p = GemmParams{};
  p.A = pb; p.Bm = vb; p.aux = x; p.outF = res1; p.outB = res1b;
  p.M = TS; p.N = TD; p.K = TS; p.lda = TS; p.ldb = TD;
  p.zDiv = 1;
  p.aOuter = (long long)TH * TS * TS; p.aSum = (long long)TS * TS;
  p.bOuter = (long long)TH * TS * TD; p.bSum = (long long)TS * TD;
  p.cOuter = (long long)TS * TD;
  p.sumCount = TH; p.kCausal = 1; p.epi = 2;
  gemm(p, TB);

  // 6) h = relu(res1 W1t + b1)
  p = GemmParams{};
  p.A = res1b; p.Bm = w1t; p.bias = b1; p.outB = hb;
  p.M = TS; p.N = TDFF; p.K = TD; p.lda = TD; p.ldb = TDFF;
  p.zDiv = 1;
  p.aOuter = (long long)TS * TD;
  p.cOuter = (long long)TS * TDFF;
  p.sumCount = 1; p.epi = 3;
  gemm(p, TB);

  // 7) out = res1 + h W2t + b2
  p = GemmParams{};
  p.A = hb; p.Bm = w2t; p.bias = b2; p.aux = res1; p.outF = out;
  p.M = TS; p.N = TD; p.K = TDFF; p.lda = TDFF; p.ldb = TD;
  p.zDiv = 1;
  p.aOuter = (long long)TS * TDFF;
  p.cOuter = (long long)TS * TD;
  p.sumCount = 1; p.epi = 4;
  gemm(p, TB);
}
